// sparse3DBA_84602265796640
// MI455X (gfx1250) — matrix operand order repaired and checked by reading, not compiled, not run on hardware
//
#include <hip/hip_runtime.h>
#include <math.h>

// ---------------------------------------------------------------------------
// Feature-metric LM pose refinement for MI455X (gfx1250, wave32).
//
// Per-sample 8-vector v = [J(6), err, 0] with
//   J_{n,c} = gx[n,c]*Mx(n) + gy[n,c]*My(n)   (M = per-point 2x6 Jacobian)
// Gram = sum_{n,c} v v^T holds Hess (6x6), g (col 6), sum err^2 ((6,6)).
// Accumulated with v_wmma_f32_16x16x32_f16: A = 16x32 (rows = v components,
// K = 32 channel-samples staged in LDS), B = A^T, f32 accumulator per wave.
// LDS slice is 16 rows x 32 K per wave; rows 7..15 are zeroed once so the
// fragment loads are unconditional ds_load_b128 (no EXEC divergence, no
// per-chunk zero-fill of the fragments).
// ---------------------------------------------------------------------------

typedef __attribute__((ext_vector_type(16))) _Float16 v16h;
typedef __attribute__((ext_vector_type(8)))  float    v8f;

#define WS_R    0    // 3x3 row-major
#define WS_T    9
#define WS_LAM  12
#define WS_LR   13
#define WS_PREV 14
#define WS_RT   16   // trial R 16..24
#define WS_TT   25   // trial t 25..27
#define WS_ACC  32   // acc[0..35]=Hess, [36..41]=g, [42]=sum e^2 cur, [43]=sum e^2 trial

__global__ void init_state(float* ws) {
  if (threadIdx.x == 0 && blockIdx.x == 0) {
    for (int i = 0; i < 9; ++i) ws[WS_R + i] = (i % 4 == 0) ? 1.f : 0.f;
    ws[WS_T + 0] = 1.f; ws[WS_T + 1] = 1.f; ws[WS_T + 2] = 0.f;
    ws[WS_LAM] = 0.01f; ws[WS_LR] = 0.1f; ws[WS_PREV] = 0.f;
  }
}

__global__ void zero_acc(float* acc) {
  int i = threadIdx.x;
  if (i < 44) acc[i] = 0.f;
}

// ---------------------------------------------------------------------------
// Main reduction: projection + gathers + WMMA Gram accumulation.
// One point per wave; lanes cover 32 channels per WMMA step (C=128 -> 4 WMMA).
// ---------------------------------------------------------------------------
__global__ __launch_bounds__(256)
void ba_reduce(const float* __restrict__ pts3D,
               const float* __restrict__ fref,
               const float* __restrict__ fmq,
               const float* __restrict__ fgx,
               const float* __restrict__ fgy,
               const float* __restrict__ Km,
               const float* __restrict__ ws,
               float* __restrict__ acc,
               int N, int C, int H, int W)
{
  __shared__ _Float16 vbuf[8 * 512];            // 8 waves x (16 rows x 32 K) halves
  const int lane = threadIdx.x & 31;
  const int wib  = threadIdx.x >> 5;
  _Float16* vb = &vbuf[wib * 512];
  const int wave   = (blockIdx.x * blockDim.x + threadIdx.x) >> 5;
  const int nwaves = (gridDim.x * blockDim.x) >> 5;

  // Zero pad rows 7..15 once; fragment loads then never need masking.
#pragma unroll
  for (int r = 7; r < 16; ++r) vb[r*32 + lane] = (_Float16)0.f;

  const float fx = Km[0], fy = Km[4], cx = Km[2], cy = Km[5];
  float R[9], t[3];
#pragma unroll
  for (int i = 0; i < 9; ++i) R[i] = ws[WS_R + i];
  t[0] = ws[WS_T + 0]; t[1] = ws[WS_T + 1]; t[2] = ws[WS_T + 2];

  const size_t plane = (size_t)H * (size_t)W;
  v8f c8 = {};                                  // per-wave 16x16 f32 Gram accumulator

  const int row = lane & 15;                    // fragment row (M for A, N for B)
  const int kbA = (lane < 16) ? 0 : 8;          // A: K {kbA..kbA+7} U {kbA+16..kbA+23}
  const int kbB = (lane < 16) ? 0 : 16;         // unused: B loads A's K set (kbA)

  for (int n = wave; n < N; n += nwaves) {
    float X = pts3D[3*n+0], Y = pts3D[3*n+1], Z = pts3D[3*n+2];
    // p = pts3D @ R^T + t
    float px = R[0]*X + R[1]*Y + R[2]*Z + t[0];
    float py = R[3]*X + R[4]*Y + R[5]*Z + t[1];
    float pz = R[6]*X + R[7]*Y + R[8]*Z + t[2];
    float invz = 1.f / pz;
    int pjx = (int)rintf((fx*px + cx*pz) * invz) - 1;
    int pjy = (int)rintf((fy*py + cy*pz) * invz) - 1;
    int jj = min(max(pjx, 0), W - 1);
    int ii = min(max(pjy, 0), H - 1);
    size_t pix = (size_t)ii * (size_t)W + (size_t)jj;

    // M = J_px_p @ [I | -skew(p)]  (2x6)
    float a = fx * invz, b = fy * invz;
    float Jx[3] = {a, 0.f, -a * px * invz};
    float Jy[3] = {0.f, b, -b * py * invz};
    float S[3][3] = {{0.f, pz, -py}, {-pz, 0.f, px}, {py, -px, 0.f}}; // -skew(p)
    float Mx[6], My[6];
#pragma unroll
    for (int k = 0; k < 3; ++k) { Mx[k] = Jx[k]; My[k] = Jy[k]; }
#pragma unroll
    for (int j = 0; j < 3; ++j) {
      float sx = 0.f, sy = 0.f;
#pragma unroll
      for (int k = 0; k < 3; ++k) { sx += Jx[k]*S[k][j]; sy += Jy[k]*S[k][j]; }
      Mx[3+j] = sx; My[3+j] = sy;
    }

    const float* frow = fref + (size_t)n * (size_t)C;

    for (int ch = 0; ch < C; ch += 32) {
      int c = ch + lane;                        // lane = K sample index
      size_t off = (size_t)c * plane + pix;
      float err = fmq[off] - frow[c];
      float gx  = fgx[off];
      float gy  = fgy[off];

      // stage v = [J(6), err] into LDS, component-major: vb[row*32 + k]
#pragma unroll
      for (int r = 0; r < 6; ++r)
        vb[r*32 + lane] = (_Float16)(gx * Mx[r] + gy * My[r]);
      vb[6*32 + lane] = (_Float16)err;

      asm volatile("s_wait_dscnt 0" ::: "memory");  // wave-local LDS RAW fence

      // A fragment (16x32 f16): lane<16 holds M=row, K {0..7}U{16..23};
      //                         lane>=16 holds M=row, K {8..15}U{24..31}
      v16h af, bf;
#pragma unroll
      for (int j = 0; j < 8; ++j) {
        af[j]     = vb[row*32 + kbA + j];
        af[8 + j] = vb[row*32 + kbA + 16 + j];
      }
      // B fragment (32x16 f16) = A^T: lane holds N=row, K {kbA..kbA+7}U{kbA+16..kbA+23}
#pragma unroll
      for (int j = 0; j < 8; ++j) {
        bf[j]     = vb[row*32 + kbA + j];
        bf[8 + j] = vb[row*32 + kbA + 16 + j];
      }

      c8 = __builtin_amdgcn_wmma_f32_16x16x32_f16(
               /*neg_a=*/false, af, /*neg_b=*/false, bf,
               /*c_mod=*/(short)0, c8, /*reuse_a=*/false, /*reuse_b=*/false);
    }
    if (n + nwaves < N)
      __builtin_prefetch(fref + (size_t)(n + nwaves) * (size_t)C, 0, 1);
  }

  // Drain wave Gram: D element (lane<16): M = vgpr r, N = lane. Useful 7x7 block.
  if (lane < 7) {
#pragma unroll
    for (int r = 0; r < 7; ++r) {
      float val = c8[r];
      if (r < 6 && lane < 6)           atomicAdd(&acc[r*6 + lane], val);   // Hess
      else if (lane == 6 && r < 6)     atomicAdd(&acc[36 + r], val);       // g
      else if (r == 6 && lane == 6)    atomicAdd(&acc[42], val);           // sum e^2
    }
  }
}

// ---------------------------------------------------------------------------
// Single-thread 6x6 solve + so3exp + trial pose.
// ---------------------------------------------------------------------------
__global__ void ba_solve(float* ws, const float* acc, int N, int it)
{
  if (threadIdx.x != 0 || blockIdx.x != 0) return;
  float A[6][7];
  float lam = ws[WS_LAM], lr = ws[WS_LR];
  for (int i = 0; i < 6; ++i) {
    for (int j = 0; j < 6; ++j) A[i][j] = acc[i*6 + j];
    A[i][6] = acc[36 + i];
  }
  float cost_cur = 0.5f * acc[42] / (float)N;
  float prev = (it == 0) ? cost_cur : ws[WS_PREV];
  ws[WS_PREV] = prev;
  for (int i = 0; i < 6; ++i) A[i][i] += (acc[i*6 + i] + 1e-9f) * lam;

  // Gauss-Jordan with partial pivoting
  for (int k = 0; k < 6; ++k) {
    int p = k; float mx = fabsf(A[k][k]);
    for (int r = k + 1; r < 6; ++r) { float v = fabsf(A[r][k]); if (v > mx) { mx = v; p = r; } }
    if (p != k) for (int j = 0; j < 7; ++j) { float tmp = A[k][j]; A[k][j] = A[p][j]; A[p][j] = tmp; }
    float inv = 1.f / A[k][k];
    for (int j = k; j < 7; ++j) A[k][j] *= inv;
    for (int r = 0; r < 6; ++r) {
      if (r == k) continue;
      float f = A[r][k];
      for (int j = k; j < 7; ++j) A[r][j] -= f * A[k][j];
    }
  }
  float dt[3], dw[3];
  for (int i = 0; i < 3; ++i) { dt[i] = -lr * A[i][6]; dw[i] = -lr * A[3 + i][6]; }

  // so3exp
  float th2 = dw[0]*dw[0] + dw[1]*dw[1] + dw[2]*dw[2];
  float theta = sqrtf(th2);
  float Aa, Bb;
  if (theta < 1e-7f) { Aa = 1.f; Bb = 0.5f; }
  else { Aa = sinf(theta) / theta; Bb = (1.f - cosf(theta)) / th2; }
  float Wm[3][3] = {{0.f, -dw[2], dw[1]}, {dw[2], 0.f, -dw[0]}, {-dw[1], dw[0], 0.f}};
  float W2[3][3];
  for (int i = 0; i < 3; ++i)
    for (int j = 0; j < 3; ++j) {
      float s = 0.f;
      for (int k = 0; k < 3; ++k) s += Wm[i][k] * Wm[k][j];
      W2[i][j] = s;
    }
  float dR[3][3];
  for (int i = 0; i < 3; ++i)
    for (int j = 0; j < 3; ++j)
      dR[i][j] = ((i == j) ? 1.f : 0.f) + Aa * Wm[i][j] + Bb * W2[i][j];

  // R_trial = dR @ R ; t_trial = dR @ t + dt
  for (int i = 0; i < 3; ++i) {
    for (int j = 0; j < 3; ++j) {
      float s = 0.f;
      for (int k = 0; k < 3; ++k) s += dR[i][k] * ws[WS_R + 3*k + j];
      ws[WS_RT + 3*i + j] = s;
    }
    float s = dt[i];
    for (int k = 0; k < 3; ++k) s += dR[i][k] * ws[WS_T + k];
    ws[WS_TT + i] = s;
  }
}

// ---------------------------------------------------------------------------
// Trial cost: gather query at trial projection, sum e^2 (gather-bound).
// ---------------------------------------------------------------------------
__global__ __launch_bounds__(256)
void ba_trial_cost(const float* __restrict__ pts3D,
                   const float* __restrict__ fref,
                   const float* __restrict__ fmq,
                   const float* __restrict__ Km,
                   const float* __restrict__ ws,
                   float* __restrict__ acc,
                   int N, int C, int H, int W)
{
  const int lane   = threadIdx.x & 31;
  const int wave   = (blockIdx.x * blockDim.x + threadIdx.x) >> 5;
  const int nwaves = (gridDim.x * blockDim.x) >> 5;
  const float fx = Km[0], fy = Km[4], cx = Km[2], cy = Km[5];
  float R[9], t[3];
#pragma unroll
  for (int i = 0; i < 9; ++i) R[i] = ws[WS_RT + i];
  t[0] = ws[WS_TT + 0]; t[1] = ws[WS_TT + 1]; t[2] = ws[WS_TT + 2];
  const size_t plane = (size_t)H * (size_t)W;
  float s = 0.f;

  for (int n = wave; n < N; n += nwaves) {
    float X = pts3D[3*n+0], Y = pts3D[3*n+1], Z = pts3D[3*n+2];
    float px = R[0]*X + R[1]*Y + R[2]*Z + t[0];
    float py = R[3]*X + R[4]*Y + R[5]*Z + t[1];
    float pz = R[6]*X + R[7]*Y + R[8]*Z + t[2];
    float invz = 1.f / pz;
    int pjx = (int)rintf((fx*px + cx*pz) * invz) - 1;
    int pjy = (int)rintf((fy*py + cy*pz) * invz) - 1;
    int jj = min(max(pjx, 0), W - 1);
    int ii = min(max(pjy, 0), H - 1);
    size_t pix = (size_t)ii * (size_t)W + (size_t)jj;
    const float* frow = fref + (size_t)n * (size_t)C;
    for (int ch = 0; ch < C; ch += 32) {
      int c = ch + lane;
      float e = fmq[(size_t)c * plane + pix] - frow[c];
      s += e * e;
    }
  }
  for (int off = 16; off > 0; off >>= 1) s += __shfl_xor(s, off, 32);
  if (lane == 0) atomicAdd(&acc[43], s);
}

__global__ void ba_accept(float* ws, const float* acc, int N)
{
  if (threadIdx.x != 0 || blockIdx.x != 0) return;
  float newc = 0.5f * acc[43] / (float)N;
  float prev = ws[WS_PREV];
  bool worse = newc > prev;
  float lam = ws[WS_LAM] * (worse ? 10.f : 0.1f);
  ws[WS_LAM] = fminf(fmaxf(lam, 1e-6f), 1e4f);
  ws[WS_LR]  = worse ? fminf(fmaxf(0.1f * ws[WS_LR], 0.001f), 1.f) : 0.1f;
  if (!worse) {
    for (int i = 0; i < 9; ++i) ws[WS_R + i] = ws[WS_RT + i];
    for (int i = 0; i < 3; ++i) ws[WS_T + i] = ws[WS_TT + i];
    ws[WS_PREV] = newc;
  }
}

__global__ void ba_writeout(const float* ws, float* out)
{
  int i = threadIdx.x;
  if (i < 12) out[i] = ws[i];   // ws[0..8]=R, ws[9..11]=t
}

// ---------------------------------------------------------------------------
extern "C" void kernel_launch(void* const* d_in, const int* in_sizes, int n_in,
                              void* d_out, int out_size, void* d_ws, size_t ws_size,
                              hipStream_t stream)
{
  const float* pts3D = (const float*)d_in[0];
  const float* fref  = (const float*)d_in[1];
  const float* fmq   = (const float*)d_in[2];
  const float* fgx   = (const float*)d_in[3];
  const float* fgy   = (const float*)d_in[4];
  const float* Km    = (const float*)d_in[5];

  const int N  = in_sizes[0] / 3;
  const int C  = in_sizes[1] / N;           // 128
  const int HW = in_sizes[2] / C;
  const int W  = 640;                       // reference setup geometry
  const int H  = HW / W;                    // 480
  const int n_iters = 5;                    // reference setup scalar

  float* ws  = (float*)d_ws;                // needs only ~80 floats
  float* acc = ws + WS_ACC;

  hipLaunchKernelGGL(init_state, dim3(1), dim3(32), 0, stream, ws);
  for (int it = 0; it < n_iters; ++it) {
    hipLaunchKernelGGL(zero_acc, dim3(1), dim3(64), 0, stream, acc);
    hipLaunchKernelGGL(ba_reduce, dim3(1024), dim3(256), 0, stream,
                       pts3D, fref, fmq, fgx, fgy, Km, ws, acc, N, C, H, W);
    hipLaunchKernelGGL(ba_solve, dim3(1), dim3(1), 0, stream, ws, acc, N, it);
    hipLaunchKernelGGL(ba_trial_cost, dim3(512), dim3(256), 0, stream,
                       pts3D, fref, fmq, Km, ws, acc, N, C, H, W);
    hipLaunchKernelGGL(ba_accept, dim3(1), dim3(1), 0, stream, ws, acc, N);
  }
  hipLaunchKernelGGL(ba_writeout, dim3(1), dim3(32), 0, stream, ws, (float*)d_out);
}
